// RelPartialLearnableMultiHeadAttn_55886114456079
// MI455X (gfx1250) — matrix operand order repaired and checked by reading, not compiled, not run on hardware
//
#include <hip/hip_runtime.h>
#include <hip/hip_bf16.h>

// ---------------- problem constants ----------------
constexpr int QLEN = 1024;
constexpr int MLEN = 1024;
constexpr int KLEN = 2048;
constexpr int BSZ  = 4;
constexpr int DM   = 1024;   // d_model
constexpr int NH   = 16;
constexpr int DH   = 64;
constexpr int ND   = NH * DH;   // 1024
constexpr int RLEN = KLEN;      // 2048

// ---------------- WMMA types ----------------
typedef __attribute__((ext_vector_type(16))) __bf16 v16bf;
typedef __attribute__((ext_vector_type(8)))  __bf16 v8bf;
typedef __attribute__((ext_vector_type(8)))  float  v8f;

union BF16Frag { v16bf v; v8bf h[2]; };

__device__ __forceinline__ unsigned short f2bf(float f) {
  union { float f; unsigned u; } v; v.f = f;
  unsigned r = (v.u + 0x7FFFu + ((v.u >> 16) & 1u)) >> 16;  // RNE
  return (unsigned short)r;
}

__device__ __forceinline__ v8f wmma_bf16(const BF16Frag& a, const BF16Frag& b, v8f c) {
  return __builtin_amdgcn_wmma_f32_16x16x32_bf16(false, a.v, false, b.v,
                                                 (short)0, c, false, false);
}

// =====================================================================
// Kernel 0: bulk f32 -> bf16 conversion (memory bound; done ONCE per
// operand so the GEMM inner loops carry zero conversion VALU).
// =====================================================================
__global__ __launch_bounds__(256)
void cvt_f32_bf16_kernel(const float* __restrict__ src,
                         unsigned short* __restrict__ dst, int n4) {
  int idx = blockIdx.x * 256 + threadIdx.x;
  if (idx < n4) {
    float4 v = ((const float4*)src)[idx];
    union { unsigned short s[4]; uint2 u; } o;
    o.s[0] = f2bf(v.x); o.s[1] = f2bf(v.y);
    o.s[2] = f2bf(v.z); o.s[3] = f2bf(v.w);
    ((uint2*)dst)[idx] = o.u;
  }
}

// =====================================================================
// Shared 32x32-per-wave bf16 GEMM core: C[32x32] += A[32xK] * B[NxK]^T
// A, B row-major bf16 with leading dims lda/ldb. 4 WMMAs per K-step.
// =====================================================================
__device__ __forceinline__ void gemm32x32_bf16(
    const unsigned short* __restrict__ A, size_t lda, int rowBase,
    const unsigned short* __restrict__ B, size_t ldb, int colBase,
    int K, int lane, v8f acc[2][2]) {
  const int ml    = lane & 15;
  const int kb    = (lane < 16) ? 0 : 8;
  const int khalf = (lane < 16) ? 0 : 16;
  const unsigned short* ar0 = A + (size_t)(rowBase + ml) * lda;
  const unsigned short* ar1 = A + (size_t)(rowBase + 16 + ml) * lda;
  const unsigned short* br0 = B + (size_t)(colBase + ml) * ldb;
  const unsigned short* br1 = B + (size_t)(colBase + 16 + ml) * ldb;
  for (int kk = 0; kk < K; kk += 32) {
    // latency-hiding prefetch two K-steps ahead (global_prefetch_b8)
    __builtin_prefetch(ar0 + kk + 64, 0, 1);
    __builtin_prefetch(ar1 + kk + 64, 0, 1);
    __builtin_prefetch(br0 + kk + 64, 0, 1);
    __builtin_prefetch(br1 + kk + 64, 0, 1);
    BF16Frag a0, a1, b0, b1;
    a0.h[0] = *(const v8bf*)(ar0 + kk + kb);
    a0.h[1] = *(const v8bf*)(ar0 + kk + 16 + kb);
    a1.h[0] = *(const v8bf*)(ar1 + kk + kb);
    a1.h[1] = *(const v8bf*)(ar1 + kk + 16 + kb);
    b0.h[0] = *(const v8bf*)(br0 + kk + kb);
    b0.h[1] = *(const v8bf*)(br0 + kk + 16 + kb);
    b1.h[0] = *(const v8bf*)(br1 + kk + kb);
    b1.h[1] = *(const v8bf*)(br1 + kk + 16 + kb);
    acc[0][0] = wmma_bf16(a0, b0, acc[0][0]);
    acc[0][1] = wmma_bf16(a0, b1, acc[0][1]);
    acc[1][0] = wmma_bf16(a1, b0, acc[1][0]);
    acc[1][1] = wmma_bf16(a1, b1, acc[1][1]);
  }
}

// =====================================================================
// Kernel 1: QKV projection  H = Xbf [8192 x 1024] @ Wqkv^T [3072 x 1024]
// Scatters q(+biases), k, v into bf16 attention layouts.
// =====================================================================
__device__ __forceinline__ void qkv_scatter(
    float val, int row, int ncol,
    const float* __restrict__ r_w_bias, const float* __restrict__ r_r_bias,
    unsigned short* __restrict__ Qa, unsigned short* __restrict__ Qb,
    unsigned short* __restrict__ Kb, unsigned short* __restrict__ Vt) {
  const int pos = row >> 2;   // position in klen
  const int b   = row & 3;    // batch
  if (ncol < ND) {                                   // q (last QLEN only)
    int hh = ncol >> 6, d = ncol & 63;
    if (pos >= MLEN) {
      int i = pos - MLEN;
      size_t o = (((size_t)(b * NH + hh)) * QLEN + i) * DH + d;
      Qa[o] = f2bf(val + r_w_bias[hh * DH + d]);
      Qb[o] = f2bf(val + r_r_bias[hh * DH + d]);
    }
  } else if (ncol < 2 * ND) {                        // k
    int cc = ncol - ND; int hh = cc >> 6, d = cc & 63;
    Kb[(((size_t)(b * NH + hh)) * KLEN + pos) * DH + d] = f2bf(val);
  } else {                                           // v (transposed [d][klen])
    int cc = ncol - 2 * ND; int hh = cc >> 6, d = cc & 63;
    Vt[(((size_t)(b * NH + hh)) * DH + d) * KLEN + pos] = f2bf(val);
  }
}

__global__ __launch_bounds__(128)
void qkv_proj_kernel(const unsigned short* __restrict__ Xbf,
                     const unsigned short* __restrict__ Wqkv,
                     const float* __restrict__ r_w_bias,
                     const float* __restrict__ r_r_bias,
                     unsigned short* __restrict__ Qa, unsigned short* __restrict__ Qb,
                     unsigned short* __restrict__ Kb, unsigned short* __restrict__ Vt) {
  const int lane    = threadIdx.x;
  const int colBase = (blockIdx.x * 4 + threadIdx.y) * 32;   // N tile (32)
  const int rowBase = blockIdx.y * 32;                       // M tile (32)
  v8f acc[2][2] = {};
  gemm32x32_bf16(Xbf, DM, rowBase, Wqkv, DM, colBase, DM, lane, acc);

  const int ml   = lane & 15;
  const int mofs = (lane < 16) ? 0 : 8;
#pragma unroll
  for (int ms = 0; ms < 2; ++ms)
#pragma unroll
    for (int ns = 0; ns < 2; ++ns)
#pragma unroll
      for (int r = 0; r < 8; ++r) {
        int row  = rowBase + ms * 16 + r + mofs;
        int ncol = colBase + ns * 16 + ml;
        qkv_scatter(acc[ms][ns][r], row, ncol, r_w_bias, r_r_bias, Qa, Qb, Kb, Vt);
      }
}

// =====================================================================
// Kernel 2: R projection  Rk[h][j][d] = Rbf [2048x1024] @ Rwbf^T
// =====================================================================
__global__ __launch_bounds__(128)
void r_proj_kernel(const unsigned short* __restrict__ Rbf,
                   const unsigned short* __restrict__ Rwbf,
                   unsigned short* __restrict__ Rk) {
  const int lane    = threadIdx.x;
  const int colBase = (blockIdx.x * 4 + threadIdx.y) * 32;
  const int rowBase = blockIdx.y * 32;
  v8f acc[2][2] = {};
  gemm32x32_bf16(Rbf, DM, rowBase, Rwbf, DM, colBase, DM, lane, acc);

  const int ml   = lane & 15;
  const int mofs = (lane < 16) ? 0 : 8;
#pragma unroll
  for (int ms = 0; ms < 2; ++ms)
#pragma unroll
    for (int ns = 0; ns < 2; ++ns)
#pragma unroll
      for (int r = 0; r < 8; ++r) {
        int row  = rowBase + ms * 16 + r + mofs;
        int ncol = colBase + ns * 16 + ml;
        int hh = ncol >> 6, d = ncol & 63;
        Rk[(((size_t)hh) * RLEN + row) * DH + d] = f2bf(acc[ms][ns][r]);
      }
}

// =====================================================================
// Kernel 3: fused flash attention with rel-shift.
// One wave per (16-query tile, b*16+h). BD via diagonal-window WMMA:
//   BD[i,j] = qb_i . Rk[1023 + j - i]  -> G = Qb_tile @ Rk[wb..wb+31]^T,
//   gather G[m][n-m+15] from 2KB LDS stage.
// =====================================================================
__global__ __launch_bounds__(32)
void flash_attn_kernel(const unsigned short* __restrict__ Qa,
                       const unsigned short* __restrict__ Qb,
                       const unsigned short* __restrict__ Kb,
                       const unsigned short* __restrict__ Vt,
                       const unsigned short* __restrict__ Rk,
                       unsigned short* __restrict__ attnV) {
  __shared__ __align__(16) float          g_stage[16][32];
  __shared__ __align__(16) unsigned short p_stage[16][32];

  const int lane  = threadIdx.x;
  const int i0    = blockIdx.x * 16;
  const int bn    = blockIdx.y;           // b*16 + h
  const int b     = bn >> 4;
  const int hh    = bn & 15;
  const int m     = lane & 15;
  const int kb    = (lane < 16) ? 0 : 8;
  const int khalf = (lane < 16) ? 0 : 16;
  const int mofs  = (lane < 16) ? 0 : 8;

  // Q fragments (K=64 -> 2 k-steps each), kept resident
  BF16Frag qa[2], qb[2];
  {
    const unsigned short* qr0 = Qa + (((size_t)bn) * QLEN + (i0 + m)) * DH;
    const unsigned short* qr1 = Qb + (((size_t)bn) * QLEN + (i0 + m)) * DH;
#pragma unroll
    for (int c = 0; c < 2; ++c) {
      qa[c].h[0] = *(const v8bf*)(qr0 + 32 * c + kb);
      qa[c].h[1] = *(const v8bf*)(qr0 + 32 * c + 16 + kb);
      qb[c].h[0] = *(const v8bf*)(qr1 + 32 * c + kb);
      qb[c].h[1] = *(const v8bf*)(qr1 + 32 * c + 16 + kb);
    }
  }

  v8f oacc[4] = {{}, {}, {}, {}};
  float mrow[8], lrow[8];
#pragma unroll
  for (int rr = 0; rr < 8; ++rr) { mrow[rr] = -1e30f; lrow[rr] = 0.f; }

  const int jmax   = MLEN + i0 + 16;           // causal bound for this tile
  const int jmax32 = (jmax + 31) & ~31;        // <= 2048 always

  for (int j0 = 0; j0 < jmax32; j0 += 32) {
#pragma unroll
    for (int sub = 0; sub < 2; ++sub) {
      const int jc = j0 + sub * 16;

      // ---- AC = Qa @ K^T (16x16) ----
      v8f sacc = {};
      {
        const unsigned short* krow = Kb + (((size_t)bn) * KLEN + (jc + m)) * DH;
#pragma unroll
        for (int c = 0; c < 2; ++c) {
          BF16Frag bfr;
          const unsigned short* p = krow + 32 * c + kb;
          bfr.h[0] = *(const v8bf*)p;
          bfr.h[1] = *(const v8bf*)(p + 16);
          sacc = wmma_bf16(qa[c], bfr, sacc);
        }
      }

      // ---- G = Qb @ Rk-window^T (16x32 diagonal window) ----
      v8f g0 = {}, g1 = {};
      const int wb = 1008 + jc - i0;           // window base in Rk rows (>=0)
      {
#pragma unroll
        for (int tsub = 0; tsub < 2; ++tsub) {
          int t = tsub * 16 + m;
          int rrow = wb + t; if (rrow > RLEN - 1) rrow = RLEN - 1;
          const unsigned short* rp = Rk + (((size_t)hh) * RLEN + rrow) * DH;
#pragma unroll
          for (int c = 0; c < 2; ++c) {
            BF16Frag bfr;
            const unsigned short* p = rp + 32 * c + kb;
            bfr.h[0] = *(const v8bf*)p;
            bfr.h[1] = *(const v8bf*)(p + 16);
            if (tsub == 0) g0 = wmma_bf16(qb[c], bfr, g0);
            else           g1 = wmma_bf16(qb[c], bfr, g1);
          }
        }
      }
#pragma unroll
      for (int rr = 0; rr < 8; ++rr) {
        g_stage[rr + mofs][m]      = g0[rr];
        g_stage[rr + mofs][16 + m] = g1[rr];
      }
      __syncthreads();

      // ---- gather BD, scale, mask, online softmax ----
#pragma unroll
      for (int rr = 0; rr < 8; ++rr) {
        const int mi = rr + mofs;
        const int iq = i0 + mi;
        const int j  = jc + m;
        float bd = g_stage[mi][m - mi + 15];   // t = (j-i)+1023-wb = n-m+15
        float s = (sacc[rr] + bd) * 0.125f;    // 1/sqrt(64)
        if (j > MLEN + iq) s = -1e30f;

        float rm = s;
        rm = fmaxf(rm, __shfl_xor(rm, 1));
        rm = fmaxf(rm, __shfl_xor(rm, 2));
        rm = fmaxf(rm, __shfl_xor(rm, 4));
        rm = fmaxf(rm, __shfl_xor(rm, 8));
        float mn   = fmaxf(mrow[rr], rm);
        float corr = __expf(mrow[rr] - mn);
        float p    = __expf(s - mn);
        float ps = p;
        ps += __shfl_xor(ps, 1);
        ps += __shfl_xor(ps, 2);
        ps += __shfl_xor(ps, 4);
        ps += __shfl_xor(ps, 8);
        lrow[rr] = lrow[rr] * corr + ps;
        mrow[rr] = mn;
        oacc[0][rr] *= corr; oacc[1][rr] *= corr;
        oacc[2][rr] *= corr; oacc[3][rr] *= corr;
        p_stage[mi][sub * 16 + m] = f2bf(p);
      }
      __syncthreads();
    }

    // ---- O += P @ V (K=32 over this j-step, 4 d-subtiles) ----
    BF16Frag pa;
    pa.h[0] = *(const v8bf*)(&p_stage[m][kb]);
    pa.h[1] = *(const v8bf*)(&p_stage[m][16 + kb]);
#pragma unroll
    for (int dt = 0; dt < 4; ++dt) {
      const unsigned short* vp =
          Vt + (((size_t)bn) * DH + dt * 16 + m) * KLEN + j0 + kb;
      BF16Frag bfr;
      bfr.h[0] = *(const v8bf*)vp;
      bfr.h[1] = *(const v8bf*)(vp + 16);
      oacc[dt] = wmma_bf16(pa, bfr, oacc[dt]);
    }
    __syncthreads();
  }

  // ---- epilogue: normalize and write attn_vec [i*BSZ+b][h*64+d] bf16 ----
#pragma unroll
  for (int rr = 0; rr < 8; ++rr) {
    const int mi = rr + mofs;
    const int iq = i0 + mi;
    const float inv = 1.0f / lrow[rr];
    size_t base = ((size_t)iq * BSZ + b) * ND + hh * DH;
#pragma unroll
    for (int dt = 0; dt < 4; ++dt) {
      attnV[base + dt * 16 + m] = f2bf(oacc[dt][rr] * inv);
    }
  }
}

// =====================================================================
// Kernel 4: output projection  attn_out = attnV [4096x1024] @ Owbf^T
// =====================================================================
__global__ __launch_bounds__(128)
void out_proj_kernel(const unsigned short* __restrict__ attnV,
                     const unsigned short* __restrict__ Owbf,
                     float* __restrict__ attnOut) {
  const int lane    = threadIdx.x;
  const int colBase = (blockIdx.x * 4 + threadIdx.y) * 32;
  const int rowBase = blockIdx.y * 32;
  v8f acc[2][2] = {};
  gemm32x32_bf16(attnV, ND, rowBase, Owbf, ND, colBase, ND, lane, acc);

  const int ml   = lane & 15;
  const int mofs = (lane < 16) ? 0 : 8;
#pragma unroll
  for (int ms = 0; ms < 2; ++ms)
#pragma unroll
    for (int ns = 0; ns < 2; ++ns)
#pragma unroll
      for (int r = 0; r < 8; ++r) {
        int row  = rowBase + ms * 16 + r + mofs;
        int ncol = colBase + ns * 16 + ml;
        attnOut[(size_t)row * DM + ncol] = acc[ms][ns][r];
      }
}

// =====================================================================
// Kernel 5: residual + LayerNorm, one block per (i,b) row
// =====================================================================
__global__ __launch_bounds__(256)
void ln_kernel(const float* __restrict__ w, const float* __restrict__ attnOut,
               const float* __restrict__ gam, const float* __restrict__ bet,
               float* __restrict__ out) {
  __shared__ float sh_s[256];
  __shared__ float sh_q[256];
  const int row = blockIdx.x;
  const int t   = threadIdx.x;
  const size_t base = (size_t)row * DM;

  float s = 0.f, q = 0.f;
  for (int c = t; c < DM; c += 256) {
    float x = w[base + c] + attnOut[base + c];
    s += x; q += x * x;
  }
  sh_s[t] = s; sh_q[t] = q;
  __syncthreads();
  for (int st = 128; st > 0; st >>= 1) {
    if (t < st) { sh_s[t] += sh_s[t + st]; sh_q[t] += sh_q[t + st]; }
    __syncthreads();
  }
  const float mu   = sh_s[0] * (1.0f / DM);
  const float var  = sh_q[0] * (1.0f / DM) - mu * mu;
  const float rinv = rsqrtf(var + 1e-5f);
  for (int c = t; c < DM; c += 256) {
    float x = w[base + c] + attnOut[base + c];
    out[base + c] = (x - mu) * rinv * gam[c] + bet[c];
  }
}

// =====================================================================
extern "C" void kernel_launch(void* const* d_in, const int* in_sizes, int n_in,
                              void* d_out, int out_size, void* d_ws, size_t ws_size,
                              hipStream_t stream) {
  const float* w        = (const float*)d_in[0];
  const float* r        = (const float*)d_in[1];
  const float* mems     = (const float*)d_in[2];
  // d_in[3] = attn_mask (causal, computed analytically in-kernel)
  const float* qkv_w    = (const float*)d_in[4];
  const float* r_w      = (const float*)d_in[5];
  const float* o_w      = (const float*)d_in[6];
  const float* r_w_bias = (const float*)d_in[7];
  const float* r_r_bias = (const float*)d_in[8];
  const float* ln_g     = (const float*)d_in[9];
  const float* ln_b     = (const float*)d_in[10];
  float* out = (float*)d_out;

  char* ws = (char*)d_ws;
  size_t off = 0;
  unsigned short* Qa   = (unsigned short*)(ws + off); off += (size_t)BSZ * NH * QLEN * DH * 2;
  unsigned short* Qb   = (unsigned short*)(ws + off); off += (size_t)BSZ * NH * QLEN * DH * 2;
  unsigned short* Kb   = (unsigned short*)(ws + off); off += (size_t)BSZ * NH * KLEN * DH * 2;
  unsigned short* Vt   = (unsigned short*)(ws + off); off += (size_t)BSZ * NH * DH * KLEN * 2;
  unsigned short* Rk   = (unsigned short*)(ws + off); off += (size_t)NH * RLEN * DH * 2;
  unsigned short* aV   = (unsigned short*)(ws + off); off += (size_t)QLEN * BSZ * ND * 2;
  float* attnOut       = (float*)(ws + off);          off += (size_t)QLEN * BSZ * DM * 4;
  unsigned short* Xbf  = (unsigned short*)(ws + off); off += (size_t)KLEN * BSZ * DM * 2;
  unsigned short* Wqkv = (unsigned short*)(ws + off); off += (size_t)3 * ND * DM * 2;
  unsigned short* Rbf  = (unsigned short*)(ws + off); off += (size_t)RLEN * DM * 2;
  unsigned short* Rwbf = (unsigned short*)(ws + off); off += (size_t)ND * DM * 2;
  unsigned short* Owbf = (unsigned short*)(ws + off); off += (size_t)DM * ND * 2;

  // ---- 0) one-shot f32->bf16 conversions (memory bound) ----
  auto cvt = [&](const float* s, unsigned short* d, size_t n) {
    int n4 = (int)(n / 4);
    cvt_f32_bf16_kernel<<<dim3((n4 + 255) / 256), dim3(256), 0, stream>>>(s, d, n4);
  };
  cvt(mems,  Xbf,                          (size_t)MLEN * BSZ * DM);  // cat rows 0..4095
  cvt(w,     Xbf + (size_t)MLEN * BSZ * DM, (size_t)QLEN * BSZ * DM); // cat rows 4096..8191
  cvt(qkv_w, Wqkv, (size_t)3 * ND * DM);
  cvt(r,     Rbf,  (size_t)RLEN * DM);
  cvt(r_w,   Rwbf, (size_t)ND * DM);
  cvt(o_w,   Owbf, (size_t)DM * ND);

  dim3 b128(32, 4, 1);
  // 1) QKV projection: 8192 x 3072 (32x32 tile per wave, 128 cols per block)
  qkv_proj_kernel<<<dim3(3072 / 128, 8192 / 32), b128, 0, stream>>>(
      Xbf, Wqkv, r_w_bias, r_r_bias, Qa, Qb, Kb, Vt);
  // 2) R projection: 2048 x 1024
  r_proj_kernel<<<dim3(1024 / 128, 2048 / 32), b128, 0, stream>>>(Rbf, Rwbf, Rk);
  // 3) fused attention: 64 q-tiles x 64 (b,h)
  flash_attn_kernel<<<dim3(QLEN / 16, BSZ * NH), dim3(32, 1, 1), 0, stream>>>(
      Qa, Qb, Kb, Vt, Rk, aV);
  // 4) output projection: 4096 x 1024
  out_proj_kernel<<<dim3(1024 / 128, 4096 / 32), b128, 0, stream>>>(aV, Owbf, attnOut);
  // 5) residual + LayerNorm
  ln_kernel<<<dim3(QLEN * BSZ), dim3(256), 0, stream>>>(w, attnOut, ln_g, ln_b, out);
}
